// mLSTMAutoEncoder_19980187861880
// MI455X (gfx1250) — matrix operand order repaired and checked by reading, not compiled, not run on hardware
//
#include <hip/hip_runtime.h>
#include <cstdint>
#include <cstddef>

// Problem constants (match reference)
#define B_ 64
#define S_ 512
#define E_ 512
#define L_ 512
#define P_ 2048

#define NBLK 32
#define NTHR 128   // 4 waves of 32; 128 waves total == 128 tile tasks per phase

typedef __attribute__((ext_vector_type(16))) __bf16 v16bf;
typedef __attribute__((ext_vector_type(8)))  __bf16 v8bf;
typedef __attribute__((ext_vector_type(8)))  float  v8f;

// ---------------------------------------------------------------------------
// helpers
// ---------------------------------------------------------------------------
static __device__ __forceinline__ v8f vzero8() {
  v8f r;
#pragma unroll
  for (int i = 0; i < 8; ++i) r[i] = 0.f;
  return r;
}

static __device__ __forceinline__ v8f wmma_bf16(v16bf a, v16bf b, v8f c) {
  // D = A(16x32 bf16) * B(32x16 bf16) + C(16x16 f32)
  return __builtin_amdgcn_wmma_f32_16x16x32_bf16(
      /*neg_a=*/false, a, /*neg_b=*/false, b,
      /*c_mod=*/(short)0, c, /*reuse_a=*/false, /*reuse_b=*/false);
}

// A fragment, 16x32 tile of row-major activation matrix, bf16 source.
// ISA layout: lanes 0-15 rows 0-15 hold K {kh..kh+7, 16+kh..16+kh+7}, kh=8*(lane>=16)
static __device__ __forceinline__ v16bf load_a(const __bf16* base, int ld,
                                               int row0, int k0) {
  const int lane = threadIdx.x & 31;
  const int row  = row0 + (lane & 15);
  const int kh   = (lane >> 4) << 3;
  const __bf16* p = base + (size_t)row * ld + k0 + kh;
  v8bf lo = *(const v8bf*)(p);        // K = k0+kh .. +7
  v8bf hi = *(const v8bf*)(p + 16);   // K = k0+16+kh .. +7
  return __builtin_shufflevector(lo, hi, 0, 1, 2, 3, 4, 5, 6, 7,
                                 8, 9, 10, 11, 12, 13, 14, 15);
}

// A fragment from f32 source (inline convert; fallback path when ws is small)
static __device__ __forceinline__ v16bf load_a(const float* base, int ld,
                                               int row0, int k0) {
  const int lane = threadIdx.x & 31;
  const int row  = row0 + (lane & 15);
  const int kh   = (lane >> 4) << 3;
  const float* p = base + (size_t)row * ld + k0 + kh;
  v16bf a;
#pragma unroll
  for (int i = 0; i < 8; ++i) a[i] = (__bf16)p[i];
#pragma unroll
  for (int i = 0; i < 8; ++i) a[8 + i] = (__bf16)p[16 + i];
  return a;
}

// B fragment, 32x16: B[k][n] = W[n0+n][k0+k] with W row-major (out,in).
// ISA layout: col = lane&15; lanes hold K {kh..kh+7, 16+kh..16+kh+7}, kh=8*(lane>=16);
// -> two 16-byte chunks of one W row per lane.
static __device__ __forceinline__ v16bf load_b_frag(const __bf16* W, int ldw,
                                                    int n0, int k0) {
  const int lane = threadIdx.x & 31;
  const int col  = lane & 15;
  const int kb   = (lane >> 4) << 3;
  const __bf16* p = W + (size_t)(n0 + col) * ldw + k0 + kb;
  v8bf lo = *(const v8bf*)(p);
  v8bf hi = *(const v8bf*)(p + 16);
  return __builtin_shufflevector(lo, hi, 0, 1, 2, 3, 4, 5, 6, 7,
                                 8, 9, 10, 11, 12, 13, 14, 15);
}

// Software grid barrier (sense via generation counter), agent scope.
static __device__ __forceinline__ void grid_barrier(unsigned* cnt, unsigned* gen,
                                                    unsigned nblk) {
  __syncthreads();
  if (threadIdx.x == 0) {
    __threadfence();
    unsigned g = __hip_atomic_load(gen, __ATOMIC_ACQUIRE, __HIP_MEMORY_SCOPE_AGENT);
    unsigned t = __hip_atomic_fetch_add(cnt, 1u, __ATOMIC_ACQ_REL, __HIP_MEMORY_SCOPE_AGENT);
    if (t == nblk - 1u) {
      __hip_atomic_store(cnt, 0u, __ATOMIC_RELAXED, __HIP_MEMORY_SCOPE_AGENT);
      __hip_atomic_fetch_add(gen, 1u, __ATOMIC_ACQ_REL, __HIP_MEMORY_SCOPE_AGENT);
    } else {
      while (__hip_atomic_load(gen, __ATOMIC_ACQUIRE, __HIP_MEMORY_SCOPE_AGENT) == g) {
        __builtin_amdgcn_s_sleep(2);
      }
    }
  }
  __syncthreads();
}

static __device__ __forceinline__ float sigmoidf_(float x) {
  return 1.f / (1.f + __expf(-x));
}

// ---------------------------------------------------------------------------
// Phase 1: m = (X @ Wmx^T) * (H @ Wmh^T)   (64 x 512), one 16x16 tile per wave
// ---------------------------------------------------------------------------
template <typename AT>
static __device__ __forceinline__ void run_phase1(
    const AT* __restrict__ xsrc, int ldx, const __bf16* __restrict__ hb,
    const __bf16* __restrict__ Wmx, const __bf16* __restrict__ Wmh,
    __bf16* __restrict__ mb, int wave) {
  const int r = wave >> 5;        // 0..3 row tile (batch)
  const int n = wave & 31;        // 0..31 col tile (latent)
  v8f accx = vzero8(), acch = vzero8();
  for (int k0 = 0; k0 < E_; k0 += 32) {
    v16bf a = load_a(xsrc, ldx, r * 16, k0);
    accx = wmma_bf16(a, load_b_frag(Wmx, E_, n * 16, k0), accx);
    v16bf ah = load_a(hb, L_, r * 16, k0);
    acch = wmma_bf16(ah, load_b_frag(Wmh, L_, n * 16, k0), acch);
  }
  const int lane = threadIdx.x & 31;
  const int col  = n * 16 + (lane & 15);
  const int rb   = r * 16 + ((lane >> 4) << 3);
#pragma unroll
  for (int v = 0; v < 8; ++v)
    mb[(size_t)(rb + v) * L_ + col] = (__bf16)(accx[v] * acch[v]);
}

// ---------------------------------------------------------------------------
// Phase 2: z = X @ Wx^T + M @ Wh^T + b ; gates ; c,h update.
// One wave computes all 4 gate tiles for hidden cols [16n,16n+16).
// WOUT (compile-time): also write f32 h to outp with row stride ostride.
// ---------------------------------------------------------------------------
template <bool WOUT, typename AT>
static __device__ __forceinline__ void run_phase2(
    const AT* __restrict__ xsrc, int ldx, const __bf16* __restrict__ mb,
    const __bf16* __restrict__ Wx, const __bf16* __restrict__ Wh,
    const float* __restrict__ bias, float* __restrict__ Cst,
    __bf16* __restrict__ hdst, float* __restrict__ outp, size_t ostride,
    int wave) {
  const int r = wave >> 5;
  const int n = wave & 31;
  v8f acc0 = vzero8(), acc1 = vzero8(), acc2 = vzero8(), acc3 = vzero8();

  for (int k0 = 0; k0 < E_; k0 += 32) {
    v16bf a = load_a(xsrc, ldx, r * 16, k0);
    acc0 = wmma_bf16(a, load_b_frag(Wx, E_, 0 * L_ + n * 16, k0), acc0);
    acc1 = wmma_bf16(a, load_b_frag(Wx, E_, 1 * L_ + n * 16, k0), acc1);
    acc2 = wmma_bf16(a, load_b_frag(Wx, E_, 2 * L_ + n * 16, k0), acc2);
    acc3 = wmma_bf16(a, load_b_frag(Wx, E_, 3 * L_ + n * 16, k0), acc3);
  }
  for (int k0 = 0; k0 < L_; k0 += 32) {
    v16bf a = load_a(mb, L_, r * 16, k0);
    acc0 = wmma_bf16(a, load_b_frag(Wh, L_, 0 * L_ + n * 16, k0), acc0);
    acc1 = wmma_bf16(a, load_b_frag(Wh, L_, 1 * L_ + n * 16, k0), acc1);
    acc2 = wmma_bf16(a, load_b_frag(Wh, L_, 2 * L_ + n * 16, k0), acc2);
    acc3 = wmma_bf16(a, load_b_frag(Wh, L_, 3 * L_ + n * 16, k0), acc3);
  }

  const int lane = threadIdx.x & 31;
  const int col  = n * 16 + (lane & 15);
  const int rb   = r * 16 + ((lane >> 4) << 3);
  const float bi = bias[0 * L_ + col];
  const float bf = bias[1 * L_ + col];
  const float bo = bias[2 * L_ + col];
  const float bu = bias[3 * L_ + col];
#pragma unroll
  for (int v = 0; v < 8; ++v) {
    const int row = rb + v;
    const size_t idx = (size_t)row * L_ + col;
    const float si = sigmoidf_(acc0[v] + bi);
    const float sf = sigmoidf_(acc1[v] + bf);
    const float so = sigmoidf_(acc2[v] + bo);
    const float tu = tanhf(acc3[v] + bu);
    const float cn = sf * Cst[idx] + si * tu;
    Cst[idx] = cn;
    const float h = so * tanhf(cn);
    hdst[idx] = (__bf16)h;
    if constexpr (WOUT) outp[(size_t)row * ostride + col] = h;
  }
}

// ---------------------------------------------------------------------------
// Persistent cooperative kernel: 512 encoder steps + 512 decoder steps.
// All phase call sites pass pointers that are directly kernel arguments so
// infer-address-spaces proves GLOBAL (no flat loads / DScnt coupling).
// XT = __bf16 -> x pre-converted (preferred). XT = float -> inline convert.
// ---------------------------------------------------------------------------
template <typename XT>
__global__ void __launch_bounds__(NTHR) mlstm_persistent(
    const XT* __restrict__ xin, const float* __restrict__ enc_b,
    const float* __restrict__ dec_b,
    const __bf16* __restrict__ eWmx, const __bf16* __restrict__ eWmh,
    const __bf16* __restrict__ eWx,  const __bf16* __restrict__ eWh,
    const __bf16* __restrict__ dWmx, const __bf16* __restrict__ dWmh,
    const __bf16* __restrict__ dWx,  const __bf16* __restrict__ dWh,
    __bf16* __restrict__ Mb, __bf16* __restrict__ H0, __bf16* __restrict__ H1,
    __bf16* __restrict__ Ib, float* __restrict__ Cst,
    float* __restrict__ out, unsigned* __restrict__ ctr) {
  const unsigned nblk = gridDim.x;
  unsigned* cnt = ctr;
  unsigned* gen = ctr + 1;
  const int wave = blockIdx.x * (blockDim.x >> 5) + (threadIdx.x >> 5);

  // ---- encoder steps 0..S-2: h,c in H0/Cst (pre-zeroed); no output ----
  for (int s = 0; s < S_ - 1; ++s) {
    const XT* xt = xin + (size_t)s * E_;                  // row stride S_*E_
    run_phase1(xt, S_ * E_, H0, eWmx, eWmh, Mb, wave);
    grid_barrier(cnt, gen, nblk);
    run_phase2<false>(xt, S_ * E_, Mb, eWx, eWh, enc_b, Cst, H0, out,
                      (size_t)L_, wave);
    grid_barrier(cnt, gen, nblk);
  }
  // ---- encoder last step: also writes latents (f32) ----
  {
    const XT* xt = xin + (size_t)(S_ - 1) * E_;
    run_phase1(xt, S_ * E_, H0, eWmx, eWmh, Mb, wave);
    grid_barrier(cnt, gen, nblk);
    run_phase2<true>(xt, S_ * E_, Mb, eWx, eWh, enc_b, Cst, H0,
                     out + (size_t)B_ * S_ * L_, (size_t)L_, wave);
    grid_barrier(cnt, gen, nblk);
  }

  // ---- transition: Ib = latents (H0); H0 = 0; C = 0 ----
  {
    const int tid = blockIdx.x * blockDim.x + threadIdx.x;
    const int nt  = gridDim.x * blockDim.x;
    for (int i = tid; i < B_ * L_; i += nt) {
      Ib[i]  = H0[i];
      H0[i]  = (__bf16)0.f;
      Cst[i] = 0.f;
    }
  }
  grid_barrier(cnt, gen, nblk);

  // ---- decoder step 0: inp = latents (Ib), h = H0 (zero), next h -> H1 ----
  run_phase1(Ib, L_, H0, dWmx, dWmh, Mb, wave);
  grid_barrier(cnt, gen, nblk);
  run_phase2<true>(Ib, L_, Mb, dWx, dWh, dec_b, Cst, H1,
                   out + (size_t)(S_ - 1) * L_, (size_t)S_ * L_, wave);
  grid_barrier(cnt, gen, nblk);

  // ---- decoder steps 1..S-1: inp == h; ping-pong H1 <-> H0.
  // Separate arms so every pointer at the call site is a direct kernel arg.
  for (int s = 1; s < S_; ++s) {
    float* op = out + (size_t)(S_ - 1 - s) * L_;
    if (s & 1) {  // h in H1, write H0
      run_phase1(H1, L_, H1, dWmx, dWmh, Mb, wave);
      grid_barrier(cnt, gen, nblk);
      run_phase2<true>(H1, L_, Mb, dWx, dWh, dec_b, Cst, H0, op,
                       (size_t)S_ * L_, wave);
      grid_barrier(cnt, gen, nblk);
    } else {      // h in H0, write H1
      run_phase1(H0, L_, H0, dWmx, dWmh, Mb, wave);
      grid_barrier(cnt, gen, nblk);
      run_phase2<true>(H0, L_, Mb, dWx, dWh, dec_b, Cst, H1, op,
                       (size_t)S_ * L_, wave);
      grid_barrier(cnt, gen, nblk);
    }
  }
}

// ---------------------------------------------------------------------------
// Weight-norm + bf16 convert: one block per output row
// ---------------------------------------------------------------------------
__global__ void wn_bf16_kernel(const float* __restrict__ v,
                               const float* __restrict__ g,
                               __bf16* __restrict__ dst, int in_dim) {
  const int row = blockIdx.x;
  const float* vr = v + (size_t)row * in_dim;
  float ss = 0.f;
  for (int i = threadIdx.x; i < in_dim; i += blockDim.x) {
    const float t = vr[i];
    ss += t * t;
  }
  __shared__ float red[256];
  red[threadIdx.x] = ss;
  __syncthreads();
  for (int off = 128; off > 0; off >>= 1) {
    if ((int)threadIdx.x < off) red[threadIdx.x] += red[threadIdx.x + off];
    __syncthreads();
  }
  const float scale = g[row] * rsqrtf(red[0]);
  for (int i = threadIdx.x; i < in_dim; i += blockDim.x)
    dst[(size_t)row * in_dim + i] = (__bf16)(vr[i] * scale);
}

// Bulk f32 -> bf16 convert (one pass over x)
__global__ void f32_to_bf16_kernel(const float* __restrict__ src,
                                   __bf16* __restrict__ dst, int n) {
  const int i = (blockIdx.x * blockDim.x + threadIdx.x) * 4;
  if (i + 3 < n) {
    const float4 v = *(const float4*)(src + i);
    dst[i + 0] = (__bf16)v.x;
    dst[i + 1] = (__bf16)v.y;
    dst[i + 2] = (__bf16)v.z;
    dst[i + 3] = (__bf16)v.w;
  } else {
    for (int j = i; j < n; ++j) dst[j] = (__bf16)src[j];
  }
}

__global__ void zero_u32_kernel(unsigned* __restrict__ p, int n) {
  const int i = blockIdx.x * blockDim.x + threadIdx.x;
  if (i < n) p[i] = 0u;
}

// ---------------------------------------------------------------------------
// host launcher
// ---------------------------------------------------------------------------
extern "C" void kernel_launch(void* const* d_in, const int* in_sizes, int n_in,
                              void* d_out, int out_size, void* d_ws, size_t ws_size,
                              hipStream_t stream) {
  (void)in_sizes; (void)n_in; (void)out_size;
  const float* x         = (const float*)d_in[0];
  const float* enc_wmx_v = (const float*)d_in[1];
  const float* enc_wmx_g = (const float*)d_in[2];
  const float* enc_wmh_v = (const float*)d_in[3];
  const float* enc_wmh_g = (const float*)d_in[4];
  const float* enc_wx_v  = (const float*)d_in[5];
  const float* enc_wx_g  = (const float*)d_in[6];
  const float* enc_wh_v  = (const float*)d_in[7];
  const float* enc_wh_g  = (const float*)d_in[8];
  const float* enc_wh_b  = (const float*)d_in[9];
  const float* dec_wmx_v = (const float*)d_in[10];
  const float* dec_wmx_g = (const float*)d_in[11];
  const float* dec_wmh_v = (const float*)d_in[12];
  const float* dec_wmh_g = (const float*)d_in[13];
  const float* dec_wx_v  = (const float*)d_in[14];
  const float* dec_wx_g  = (const float*)d_in[15];
  const float* dec_wh_v  = (const float*)d_in[16];
  const float* dec_wh_g  = (const float*)d_in[17];
  const float* dec_wh_b  = (const float*)d_in[18];
  float* out = (float*)d_out;

  // scratch carve-out (256B aligned regions)
  char* ws = (char*)d_ws;
  size_t off = 0;
  auto carve = [&](size_t bytes) -> void* {
    void* p = ws + off;
    off += (bytes + 255) & ~(size_t)255;
    return p;
  };
  unsigned* ctr = (unsigned*)carve(256);
  __bf16* eWmx = (__bf16*)carve((size_t)L_ * E_ * 2);
  __bf16* eWmh = (__bf16*)carve((size_t)L_ * L_ * 2);
  __bf16* eWx  = (__bf16*)carve((size_t)P_ * E_ * 2);
  __bf16* eWh  = (__bf16*)carve((size_t)P_ * L_ * 2);
  __bf16* dWmx = (__bf16*)carve((size_t)L_ * E_ * 2);
  __bf16* dWmh = (__bf16*)carve((size_t)L_ * L_ * 2);
  __bf16* dWx  = (__bf16*)carve((size_t)P_ * E_ * 2);
  __bf16* dWh  = (__bf16*)carve((size_t)P_ * L_ * 2);
  __bf16* Mb   = (__bf16*)carve((size_t)B_ * L_ * 2);
  __bf16* H0   = (__bf16*)carve((size_t)B_ * L_ * 2);
  __bf16* H1   = (__bf16*)carve((size_t)B_ * L_ * 2);
  __bf16* Ib   = (__bf16*)carve((size_t)B_ * L_ * 2);
  float*  Cst  = (float*)carve((size_t)B_ * L_ * 4);
  __bf16* Xb   = (__bf16*)carve((size_t)B_ * S_ * E_ * 2);  // optional bf16 x
  const bool use_xbf = (ws_size >= off);                    // deterministic in ws_size

  // init state: barrier counters, h0 = 0, c = 0
  zero_u32_kernel<<<1, 64, 0, stream>>>(ctr, 2);
  zero_u32_kernel<<<(B_ * L_ / 2 + 255) / 256, 256, 0, stream>>>((unsigned*)H0,
                                                                 B_ * L_ / 2);
  zero_u32_kernel<<<(B_ * L_ + 255) / 256, 256, 0, stream>>>((unsigned*)Cst,
                                                             B_ * L_);
  // weight-norm + bf16 (loop-invariant, once per call)
  wn_bf16_kernel<<<L_, 256, 0, stream>>>(enc_wmx_v, enc_wmx_g, eWmx, E_);
  wn_bf16_kernel<<<L_, 256, 0, stream>>>(enc_wmh_v, enc_wmh_g, eWmh, L_);
  wn_bf16_kernel<<<P_, 256, 0, stream>>>(enc_wx_v,  enc_wx_g,  eWx,  E_);
  wn_bf16_kernel<<<P_, 256, 0, stream>>>(enc_wh_v,  enc_wh_g,  eWh,  L_);
  wn_bf16_kernel<<<L_, 256, 0, stream>>>(dec_wmx_v, dec_wmx_g, dWmx, E_);
  wn_bf16_kernel<<<L_, 256, 0, stream>>>(dec_wmh_v, dec_wmh_g, dWmh, L_);
  wn_bf16_kernel<<<P_, 256, 0, stream>>>(dec_wx_v,  dec_wx_g,  dWx,  E_);
  wn_bf16_kernel<<<P_, 256, 0, stream>>>(dec_wh_v,  dec_wh_g,  dWh,  L_);

  if (use_xbf) {
    const int nx = B_ * S_ * E_;
    f32_to_bf16_kernel<<<(nx / 4 + 255) / 256, 256, 0, stream>>>(x, Xb, nx);
    mlstm_persistent<__bf16><<<NBLK, NTHR, 0, stream>>>(
        Xb, enc_wh_b, dec_wh_b, eWmx, eWmh, eWx, eWh,
        dWmx, dWmh, dWx, dWh, Mb, H0, H1, Ib, Cst, out, ctr);
  } else {
    mlstm_persistent<float><<<NBLK, NTHR, 0, stream>>>(
        x, enc_wh_b, dec_wh_b, eWmx, eWmh, eWx, eWh,
        dWmx, dWmh, dWx, dWh, Mb, H0, H1, Ib, Cst, out, ctr);
  }
}
